// Attention_22393959481545
// MI455X (gfx1250) — matrix operand order repaired and checked by reading, not compiled, not run on hardware
//
#include <hip/hip_runtime.h>
#include <hip/hip_bf16.h>
#include <stdint.h>

// ---------------- problem constants ----------------
#define BATCH   2
#define NSEQ    2048
#define DMODEL  1024
#define HEADS   16
#define DHEAD   64
#define INNER   1024            // HEADS*DHEAD
#define MTOT    (BATCH*NSEQ)    // 4096
#define QKV_N   (3*INNER)       // 3072
#define QSCALE  0.125f          // DHEAD^-0.5
#define LNEPS   1e-5f

typedef __attribute__((ext_vector_type(16))) __bf16 v16bf;
typedef __attribute__((ext_vector_type(8)))  __bf16 v8bf;
typedef __attribute__((ext_vector_type(8)))  float  v8f;
typedef __attribute__((ext_vector_type(4)))  unsigned int u32x4;
typedef __attribute__((ext_vector_type(8)))  int i32x8;
typedef __attribute__((ext_vector_type(4)))  int i32x4;

// TDM builtin exists on both ROCm 7.2 (clang-22, 5-arg) and amdgpu-toolchain
// (clang-23, 6-arg); dispatch on clang major for the arity.
#if defined(__HIP_DEVICE_COMPILE__) && \
    __has_builtin(__builtin_amdgcn_tensor_load_to_lds) && \
    __has_builtin(__builtin_amdgcn_s_wait_tensorcnt)
#define USE_TDM 1
#else
#define USE_TDM 0
#endif

// ---------------- helpers ----------------
__device__ __forceinline__ v8f vzero8() {
  v8f z = {0.f,0.f,0.f,0.f,0.f,0.f,0.f,0.f};
  return z;
}

// A-fragment (16x32 bf16, row-major LDS tile, row stride ld elements).
// ISA layout: lanes 0-15 row M=lane, K={0..7,16..23}; lanes 16-31 row M=lane-16,
// K={8..15,24..31}.  Two ds_load_b128 per lane.
__device__ __forceinline__ v16bf load_frag_a(const __bf16* base, int ld,
                                             int m0, int k0, int lane) {
  int l15 = lane & 15, hl = lane >> 4;
  const __bf16* p = base + (size_t)(m0 + l15) * ld + k0 + hl * 8;
  v8bf lo = *(const v8bf*)p;
  v8bf hi = *(const v8bf*)(p + 16);
  v16bf a;
#pragma unroll
  for (int e = 0; e < 8; ++e) { a[e] = lo[e]; a[8 + e] = hi[e]; }
  return a;
}

// B-fragment (32x16 bf16) from K-transposed LDS tile Bs[n][k], row stride ld.
// ISA layout: lanes 0-15 K={0..7,16..23}, lanes 16-31 K={8..15,24..31}, N=lane&15.
__device__ __forceinline__ v16bf load_frag_b(const __bf16* base, int ld,
                                             int n0, int k0, int lane) {
  int l15 = lane & 15, hl = lane >> 4;
  const __bf16* p = base + (size_t)(n0 + l15) * ld + k0 + hl * 8;
  v8bf lo = *(const v8bf*)p;
  v8bf hi = *(const v8bf*)(p + 16);
  v16bf b;
#pragma unroll
  for (int e = 0; e < 8; ++e) { b[e] = lo[e]; b[8 + e] = hi[e]; }
  return b;
}

#if USE_TDM
// TDM 2D tile load: `rows` x 64 bf16 tile (row length 64 elements = 128B),
// global row stride `row_stride` elements, into LDS at lds_addr_b with 16B
// padding appended after every 128B row (reproduces the 72-element padded
// LDS row layout).  D# bit packing per CDNA5 ISA §8.3/§8.4.
__device__ __forceinline__ void tdm_load_tile(unsigned lds_addr_b,
                                              const __bf16* gaddr,
                                              unsigned rows,
                                              unsigned row_stride) {
  unsigned long long ga = (unsigned long long)(uintptr_t)gaddr;
  u32x4 g0;
  g0[0] = 1u;                                    // count=1, user descriptor
  g0[1] = lds_addr_b;                            // lds_addr
  g0[2] = (unsigned)(ga & 0xffffffffu);          // global_addr[31:0]
  g0[3] = (unsigned)((ga >> 32) & 0x1ffffffu)    // global_addr[56:32]
        | (2u << 30);                            // type=2 ("image")
  i32x8 g1;
  g1[0] = (int)((1u << 16)                       // data_size = 2 bytes
              | (1u << 20)                       // pad_enable
              | (4u << 22)                       // pad_interval: 32 DWORDs (128B)
              | (3u << 25));                     // pad_amount: 4 DWORDs (16B)
  g1[1] = (int)(64u << 16);                      // tensor_dim0[15:0] = 64
  g1[2] = (int)((rows & 0xffffu) << 16);         // dim0 hi=0 | tensor_dim1 lo
  g1[3] = (int)((rows >> 16) | (64u << 16));     // dim1 hi | tile_dim0=64
  g1[4] = (int)(rows & 0xffffu);                 // tile_dim1=rows, tile_dim2=0
  g1[5] = (int)row_stride;                       // tensor_dim0_stride[31:0]
  g1[6] = 0;                                     // stride hi | dim1_stride lo
  g1[7] = 0;
  i32x4 z4 = {0, 0, 0, 0};
#if __clang_major__ >= 23
  i32x8 z8 = {0, 0, 0, 0, 0, 0, 0, 0};
  __builtin_amdgcn_tensor_load_to_lds(g0, g1, z4, z4, z8, 0);   // 6-arg form
#else
  __builtin_amdgcn_tensor_load_to_lds(g0, g1, z4, z4, 0);       // 5-arg form
#endif
}
__device__ __forceinline__ unsigned lds_byte_addr(const void* p) {
  return (unsigned)(uintptr_t)p;   // generic LDS addr: low 32 bits = LDS offset
}
#endif

// ---------------- kernel 1: f32 -> bf16 transpose-convert ----------------
// src [K][N] f32  ->  dst [N][K] bf16   (LDS-tiled, coalesced both sides)
__global__ __launch_bounds__(256)
void transpose_cvt_kernel(const float* __restrict__ src,
                          __bf16* __restrict__ dst, int K, int N) {
  __shared__ float tile[32][33];
  int cx = threadIdx.x & 31;       // fast index
  int ry = threadIdx.x >> 5;       // 0..7
  int k0 = blockIdx.x * 32, n0 = blockIdx.y * 32;
#pragma unroll
  for (int i = 0; i < 4; ++i)
    tile[ry + i * 8][cx] = src[(size_t)(k0 + ry + i * 8) * N + n0 + cx];
  __syncthreads();
#pragma unroll
  for (int i = 0; i < 4; ++i)
    dst[(size_t)(n0 + ry + i * 8) * K + k0 + cx] = (__bf16)tile[cx][ry + i * 8];
}

// ---------------- kernel 2: LayerNorm -> bf16 ----------------
__global__ __launch_bounds__(256)
void layernorm_kernel(const float* __restrict__ x,
                      const float* __restrict__ gamma,
                      const float* __restrict__ beta,
                      __bf16* __restrict__ xln) {
  __shared__ float rsum[8], rsq[8];
  int row  = blockIdx.x;
  int tid  = threadIdx.x;
  int lane = tid & 31, wave = tid >> 5;
  const float4* xr = (const float4*)(x + (size_t)row * DMODEL);
  float4 v = xr[tid];
  float s  = v.x + v.y + v.z + v.w;
  float ss = v.x*v.x + v.y*v.y + v.z*v.z + v.w*v.w;
#pragma unroll
  for (int off = 16; off >= 1; off >>= 1) {
    s  += __shfl_xor(s,  off, 32);
    ss += __shfl_xor(ss, off, 32);
  }
  if (lane == 0) { rsum[wave] = s; rsq[wave] = ss; }
  __syncthreads();
  float st = 0.f, sst = 0.f;
#pragma unroll
  for (int i = 0; i < 8; ++i) { st += rsum[i]; sst += rsq[i]; }
  float mean = st * (1.0f / DMODEL);
  float var  = sst * (1.0f / DMODEL) - mean * mean;
  float rstd = rsqrtf(var + LNEPS);
  int c = tid * 4;
  float4 g = ((const float4*)gamma)[tid];
  float4 bb = ((const float4*)beta)[tid];
  __bf16* o = xln + (size_t)row * DMODEL + c;
  o[0] = (__bf16)((v.x - mean) * rstd * g.x + bb.x);
  o[1] = (__bf16)((v.y - mean) * rstd * g.y + bb.y);
  o[2] = (__bf16)((v.z - mean) * rstd * g.z + bb.z);
  o[3] = (__bf16)((v.w - mean) * rstd * g.w + bb.w);
}

// ---------------- GEMM tiling ----------------
#define BM 128
#define BN 128
#define BK 64
#define LDT 72   // BK + 8 pad: 144B rows, 16B-aligned fragment offsets

// One BK=64 stage of WMMA work for a wave (4x2 16x16 tiles).
__device__ __forceinline__ void stage_compute(const __bf16* As, const __bf16* Bs,
                                              v8f acc[4][2], int wm, int wn,
                                              int lane) {
#pragma unroll
  for (int kk = 0; kk < 2; ++kk) {
    v16bf af[4], bfv[2];
#pragma unroll
    for (int tm = 0; tm < 4; ++tm)
      af[tm] = load_frag_a(As, LDT, wm * 64 + tm * 16, kk * 32, lane);
#pragma unroll
    for (int tn = 0; tn < 2; ++tn)
      bfv[tn] = load_frag_b(Bs, LDT, wn * 32 + tn * 16, kk * 32, lane);
#pragma unroll
    for (int tm = 0; tm < 4; ++tm)
#pragma unroll
      for (int tn = 0; tn < 2; ++tn)
        acc[tm][tn] = __builtin_amdgcn_wmma_f32_16x16x32_bf16(
            false, af[tm], false, bfv[tn], (short)0, acc[tm][tn], false, false);
  }
}

#if USE_TDM
// TDM-staged, double-buffered GEMM core.  Wave 0 issues two TENSOR_LOAD_TO_LDS
// per k-step (A tile 128x64, B tile 128x64) for the *next* stage, then waits
// tensorcnt<=2 so the current stage's tiles are resident before the barrier.
#define GEMM_SMEM                                                              \
  __shared__ __align__(16) __bf16 As[2][BM * LDT];                             \
  __shared__ __align__(16) __bf16 Bs[2][BN * LDT];
#define GEMM_CORE(A_, Wt_, Kdim_)                                              \
  v8f acc[4][2];                                                               \
  _Pragma("unroll") for (int i = 0; i < 4; ++i)                                \
      _Pragma("unroll") for (int jj = 0; jj < 2; ++jj) acc[i][jj] = vzero8();  \
  if (wave == 0) {                                                             \
    tdm_load_tile(lds_byte_addr(&As[0][0]), &(A_)[(size_t)m0 * (Kdim_)], BM, (Kdim_)); \
    tdm_load_tile(lds_byte_addr(&Bs[0][0]), &(Wt_)[(size_t)n0 * (Kdim_)], BN, (Kdim_)); \
  }                                                                            \
  for (int kt = 0; kt < (Kdim_) / BK; ++kt) {                                  \
    if (wave == 0) {                                                           \
      if (kt + 1 < (Kdim_) / BK) {                                             \
        tdm_load_tile(lds_byte_addr(&As[(kt + 1) & 1][0]),                     \
                      &(A_)[(size_t)m0 * (Kdim_) + (kt + 1) * BK], BM, (Kdim_)); \
        tdm_load_tile(lds_byte_addr(&Bs[(kt + 1) & 1][0]),                     \
                      &(Wt_)[(size_t)n0 * (Kdim_) + (kt + 1) * BK], BN, (Kdim_)); \
        __builtin_amdgcn_s_wait_tensorcnt(2);                                  \
      } else {                                                                 \
        __builtin_amdgcn_s_wait_tensorcnt(0);                                  \
      }                                                                        \
    }                                                                          \
    __syncthreads();                                                           \
    stage_compute(&As[kt & 1][0], &Bs[kt & 1][0], acc, wm, wn, lane);          \
    __syncthreads();                                                           \
  }
#else
// Fallback: vector-load staging; batch all global loads before LDS stores.
#define GEMM_SMEM                                                              \
  __shared__ __align__(16) __bf16 As[1][BM * LDT];                             \
  __shared__ __align__(16) __bf16 Bs[1][BN * LDT];
#define GEMM_CORE(A_, Wt_, Kdim_)                                              \
  v8f acc[4][2];                                                               \
  _Pragma("unroll") for (int i = 0; i < 4; ++i)                                \
      _Pragma("unroll") for (int jj = 0; jj < 2; ++jj) acc[i][jj] = vzero8();  \
  for (int kt = 0; kt < (Kdim_) / BK; ++kt) {                                  \
    __syncthreads();                                                           \
    int4 ra[4], rb[4];                                                         \
    _Pragma("unroll") for (int i = 0; i < 4; ++i) {                            \
      int cid = tid + 256 * i;                                                 \
      int row = cid >> 3, kc = cid & 7;                                        \
      ra[i] = *(const int4*)(&(A_)[(size_t)(m0 + row) * (Kdim_) + kt * BK + kc * 8]); \
      rb[i] = *(const int4*)(&(Wt_)[(size_t)(n0 + row) * (Kdim_) + kt * BK + kc * 8]); \
    }                                                                          \
    _Pragma("unroll") for (int i = 0; i < 4; ++i) {                            \
      int cid = tid + 256 * i;                                                 \
      int row = cid >> 3, kc = cid & 7;                                        \
      *(int4*)(&As[0][row * LDT + kc * 8]) = ra[i];                            \
      *(int4*)(&Bs[0][row * LDT + kc * 8]) = rb[i];                            \
    }                                                                          \
    if (kt + 1 < (Kdim_) / BK) {                                               \
      __builtin_prefetch(&(A_)[(size_t)(m0 + (tid >> 3)) * (Kdim_) + (kt + 1) * BK], 0, 0); \
      __builtin_prefetch(&(Wt_)[(size_t)(n0 + (tid >> 3)) * (Kdim_) + (kt + 1) * BK], 0, 0); \
    }                                                                          \
    __syncthreads();                                                           \
    stage_compute(&As[0][0], &Bs[0][0], acc, wm, wn, lane);                    \
  }
#endif

// kernel 3: qkv = xln @ w_qkv ; scatter into Q(scaled)/K [B,H,N,D] and
// V transposed [B,H,D,N] (so flash V^T tiles are contiguous rows).
__global__ __launch_bounds__(256)
void gemm_qkv_kernel(const __bf16* __restrict__ A,    // [4096,1024]
                     const __bf16* __restrict__ Wt,   // [3072,1024] = w_qkv^T
                     __bf16* __restrict__ Qo, __bf16* __restrict__ Ko,
                     __bf16* __restrict__ Vo) {
  GEMM_SMEM
  int tid = threadIdx.x, lane = tid & 31, wave = tid >> 5;
  int wm = wave >> 2, wn = wave & 3;
  int m0 = blockIdx.y * BM, n0 = blockIdx.x * BN;
  int l15 = lane & 15, hl = lane >> 4;

  GEMM_CORE(A, Wt, DMODEL)

#pragma unroll
  for (int tm = 0; tm < 4; ++tm)
#pragma unroll
    for (int tn = 0; tn < 2; ++tn)
#pragma unroll
      for (int r = 0; r < 8; ++r) {
        int m = m0 + wm * 64 + tm * 16 + r + 8 * hl;
        int c = n0 + wn * 32 + tn * 16 + l15;
        int which = c >> 10;            // 0=q 1=k 2=v
        int inner = c & 1023;
        int head = inner >> 6, d = inner & 63;
        int b = m >> 11, nn = m & 2047;
        float v = acc[tm][tn][r];
        size_t bhh = (size_t)(b * HEADS + head);
        if (which == 0)      Qo[(bhh * NSEQ + nn) * DHEAD + d] = (__bf16)(v * QSCALE);
        else if (which == 1) Ko[(bhh * NSEQ + nn) * DHEAD + d] = (__bf16)v;
        else                 Vo[(bhh * DHEAD + d) * NSEQ + nn] = (__bf16)v;  // transposed
      }
}

// ---------------- kernel 4: flash attention ----------------
// grid (B*H, NSEQ/64), block 128 (4 waves); wave owns 16 query rows.
// K/V tiles staged by the Tensor Data Mover (double-buffered) when available.
#define FLD 72   // 64 + 8 pad
__global__ __launch_bounds__(128)
void flash_kernel(const __bf16* __restrict__ Q, const __bf16* __restrict__ K,
                  const __bf16* __restrict__ Vt_g, __bf16* __restrict__ O) {
#if USE_TDM
  __shared__ __align__(16) __bf16 KsB[2][64 * FLD];
  __shared__ __align__(16) __bf16 VtB[2][64 * FLD];
#else
  __shared__ __align__(16) __bf16 KsB[1][64 * FLD];
  __shared__ __align__(16) __bf16 VtB[1][64 * FLD];
#endif
  __shared__ __align__(16) __bf16 Qs[64 * FLD];
  __shared__ __align__(16) __bf16 Ps[4 * 16 * FLD];
  int tid = threadIdx.x, lane = tid & 31, wave = tid >> 5;
  int l15 = lane & 15, hl = lane >> 4;
  int bh = blockIdx.x, qt = blockIdx.y;
  int b = bh >> 4, h = bh & 15;
  const __bf16* qg = Q + ((size_t)bh * NSEQ + qt * 64) * DHEAD;
  const __bf16* kg = K + (size_t)bh * NSEQ * DHEAD;
  const __bf16* vg = Vt_g + (size_t)bh * DHEAD * NSEQ;   // [DHEAD][NSEQ]

#if USE_TDM
  if (wave == 0) {
    tdm_load_tile(lds_byte_addr(&Qs[0]), qg, 64, DHEAD);          // Q tile
    tdm_load_tile(lds_byte_addr(&KsB[0][0]), kg, 64, DHEAD);      // K tile j=0
    tdm_load_tile(lds_byte_addr(&VtB[0][0]), vg, 64, NSEQ);       // V^T tile j=0
  }
#else
  { // manual Q tile
    int row = tid >> 1, d0 = (tid & 1) * 32;
#pragma unroll
    for (int i = 0; i < 4; ++i)
      *(int4*)(&Qs[row * FLD + d0 + i * 8]) =
          *(const int4*)(&qg[row * DHEAD + d0 + i * 8]);
  }
#endif

  float mstat[8], lstat[8];
  v8f accO[4];
#pragma unroll
  for (int r = 0; r < 8; ++r) { mstat[r] = -1e30f; lstat[r] = 0.f; }
#pragma unroll
  for (int dt = 0; dt < 4; ++dt) accO[dt] = vzero8();

  for (int j = 0; j < NSEQ / 64; ++j) {
#if USE_TDM
    const __bf16* Ks = &KsB[j & 1][0];
    const __bf16* Vt = &VtB[j & 1][0];
    if (wave == 0) {
      if (j + 1 < NSEQ / 64) {   // prefetch next tiles into the other buffer
        tdm_load_tile(lds_byte_addr(&KsB[(j + 1) & 1][0]),
                      kg + (size_t)(j + 1) * 64 * DHEAD, 64, DHEAD);
        tdm_load_tile(lds_byte_addr(&VtB[(j + 1) & 1][0]),
                      vg + (size_t)(j + 1) * 64, 64, NSEQ);
        __builtin_amdgcn_s_wait_tensorcnt(2);  // all but the 2 newest done
      } else {
        __builtin_amdgcn_s_wait_tensorcnt(0);
      }
    }
    __syncthreads();             // publish buf[j&1] to all waves
#else
    const __bf16* Ks = &KsB[0][0];
    const __bf16* Vt = &VtB[0][0];
    __syncthreads();             // prior-iteration consumers done
    {
      int row = tid >> 1, c0 = (tid & 1) * 32;
      int4 rk[4], rv[4];
#pragma unroll
      for (int i = 0; i < 4; ++i) {
        rk[i] = *(const int4*)(&kg[(size_t)(j * 64 + row) * DHEAD + c0 + i * 8]);
        rv[i] = *(const int4*)(&vg[(size_t)row * NSEQ + j * 64 + c0 + i * 8]);
      }
#pragma unroll
      for (int i = 0; i < 4; ++i) {
        *(int4*)(&KsB[0][row * FLD + c0 + i * 8]) = rk[i];
        *(int4*)(&VtB[0][row * FLD + c0 + i * 8]) = rv[i];
      }
    }
    __syncthreads();
#endif

    // S = Q @ K^T   (Q pre-scaled by d^-0.5)
    v16bf aq[2];
#pragma unroll
    for (int kk = 0; kk < 2; ++kk)
      aq[kk] = load_frag_a(Qs, FLD, wave * 16, kk * 32, lane);
    v8f s[4];
#pragma unroll
    for (int nt = 0; nt < 4; ++nt) {
      s[nt] = vzero8();
#pragma unroll
      for (int kk = 0; kk < 2; ++kk) {
        v16bf bk = load_frag_b(Ks, FLD, nt * 16, kk * 32, lane);
        s[nt] = __builtin_amdgcn_wmma_f32_16x16x32_bf16(
            false, aq[kk], false, bk, (short)0, s[nt], false, false);
      }
    }

    // online softmax (rows live across 16-lane halves of the wave)
#pragma unroll
    for (int r = 0; r < 8; ++r) {
      float mx = fmaxf(fmaxf(s[0][r], s[1][r]), fmaxf(s[2][r], s[3][r]));
#pragma unroll
      for (int off = 8; off >= 1; off >>= 1)
        mx = fmaxf(mx, __shfl_xor(mx, off, 16));
      float mn = fmaxf(mstat[r], mx);
      float corr = __expf(mstat[r] - mn);
      float rs = 0.f;
#pragma unroll
      for (int nt = 0; nt < 4; ++nt) {
        float p = __expf(s[nt][r] - mn);
        s[nt][r] = p;
        rs += p;
      }
#pragma unroll
      for (int off = 8; off >= 1; off >>= 1) rs += __shfl_xor(rs, off, 16);
      lstat[r] = lstat[r] * corr + rs;
      mstat[r] = mn;
#pragma unroll
      for (int dt = 0; dt < 4; ++dt) accO[dt][r] *= corr;
    }

    // P -> LDS (per-wave region), then P @ V
#pragma unroll
    for (int r = 0; r < 8; ++r)
#pragma unroll
      for (int nt = 0; nt < 4; ++nt)
        Ps[(wave * 16 + r + 8 * hl) * FLD + nt * 16 + l15] = (__bf16)s[nt][r];
    __syncthreads();

    v16bf ap[2];
#pragma unroll
    for (int kk = 0; kk < 2; ++kk)
      ap[kk] = load_frag_a(Ps, FLD, wave * 16, kk * 32, lane);
#pragma unroll
    for (int dt = 0; dt < 4; ++dt)
#pragma unroll
      for (int kk = 0; kk < 2; ++kk) {
        v16bf bv = load_frag_b(Vt, FLD, dt * 16, kk * 32, lane);
        accO[dt] = __builtin_amdgcn_wmma_f32_16x16x32_bf16(
            false, ap[kk], false, bv, (short)0, accO[dt], false, false);
      }
    __syncthreads();   // everyone done with buf[j&1]; safe to overwrite later
  }

  // write O as [B, N, H*D] bf16 for the out-proj GEMM
#pragma unroll
  for (int dt = 0; dt < 4; ++dt)
#pragma unroll
    for (int r = 0; r < 8; ++r) {
      int row = qt * 64 + wave * 16 + r + 8 * hl;
      int col = h * DHEAD + dt * 16 + l15;
      float ov = accO[dt][r] / lstat[r];
      O[((size_t)b * NSEQ + row) * INNER + col] = (__bf16)ov;
    }
}

// ---------------- kernel 5: out = O @ w_out + b_out (f32) ----------------
__global__ __launch_bounds__(256)
void gemm_out_kernel(const __bf16* __restrict__ A,    // [4096,1024]
                     const __bf16* __restrict__ Wt,   // [1024,1024] = w_out^T
                     const float* __restrict__ bias,
                     float* __restrict__ out) {
  GEMM_SMEM
  int tid = threadIdx.x, lane = tid & 31, wave = tid >> 5;
  int wm = wave >> 2, wn = wave & 3;
  int m0 = blockIdx.y * BM, n0 = blockIdx.x * BN;
  int l15 = lane & 15, hl = lane >> 4;

  GEMM_CORE(A, Wt, INNER)

#pragma unroll
  for (int tn = 0; tn < 2; ++tn) {
    int n = n0 + wn * 32 + tn * 16 + l15;
    float bv = bias[n];
#pragma unroll
    for (int tm = 0; tm < 4; ++tm)
#pragma unroll
      for (int r = 0; r < 8; ++r) {
        int m = m0 + wm * 64 + tm * 16 + r + 8 * hl;
        out[(size_t)m * DMODEL + n] = acc[tm][tn][r] + bv;
      }
  }
}

// ---------------- host launcher ----------------
extern "C" void kernel_launch(void* const* d_in, const int* in_sizes, int n_in,
                              void* d_out, int out_size, void* d_ws, size_t ws_size,
                              hipStream_t stream) {
  const float* x     = (const float*)d_in[0];
  const float* gamma = (const float*)d_in[1];
  const float* beta  = (const float*)d_in[2];
  const float* wqkv  = (const float*)d_in[3];
  const float* wout  = (const float*)d_in[4];
  const float* bout  = (const float*)d_in[5];
  float* out = (float*)d_out;

  char* ws = (char*)d_ws;
  size_t off = 0;
  auto alloc = [&](size_t bytes) {
    void* p = ws + off;
    off += (bytes + 255) & ~(size_t)255;
    return p;
  };
  __bf16* xln    = (__bf16*)alloc((size_t)MTOT * DMODEL * 2);
  __bf16* wqkvT  = (__bf16*)alloc((size_t)QKV_N * DMODEL * 2);  // [3072][1024]
  __bf16* woutT  = (__bf16*)alloc((size_t)DMODEL * INNER * 2);  // [1024][1024]
  __bf16* Qb     = (__bf16*)alloc((size_t)BATCH * HEADS * NSEQ * DHEAD * 2);
  __bf16* Kb     = (__bf16*)alloc((size_t)BATCH * HEADS * NSEQ * DHEAD * 2);
  __bf16* Vb     = (__bf16*)alloc((size_t)BATCH * HEADS * DHEAD * NSEQ * 2);
  __bf16* Ob     = (__bf16*)alloc((size_t)MTOT * INNER * 2);

  // w_qkv [1024][3072] -> wqkvT [3072][1024]; w_out [1024][1024] -> woutT
  transpose_cvt_kernel<<<dim3(DMODEL / 32, QKV_N / 32), 256, 0, stream>>>(
      wqkv, wqkvT, DMODEL, QKV_N);
  transpose_cvt_kernel<<<dim3(INNER / 32, DMODEL / 32), 256, 0, stream>>>(
      wout, woutT, INNER, DMODEL);
  layernorm_kernel<<<MTOT, 256, 0, stream>>>(x, gamma, beta, xln);
  gemm_qkv_kernel<<<dim3(QKV_N / BN, MTOT / BM), 256, 0, stream>>>(
      xln, wqkvT, Qb, Kb, Vb);
  flash_kernel<<<dim3(BATCH * HEADS, NSEQ / 64), 128, 0, stream>>>(
      Qb, Kb, Vb, Ob);
  gemm_out_kernel<<<dim3(DMODEL / BN, MTOT / BM), 256, 0, stream>>>(
      Ob, woutT, bout, out);
}
